// SynapseNet_GAT_PointNet2Transformer_with_attr_4037269258365
// MI455X (gfx1250) — compile-verified, matrix operand order checked by reading, not run on hardware
//
#include <hip/hip_runtime.h>
#include <hip/hip_bf16.h>
#include <cstdint>
#include <cstddef>

typedef _Float16 h16;
typedef __attribute__((ext_vector_type(16))) _Float16 v16h;
typedef __attribute__((ext_vector_type(8)))  _Float16 v8h;
typedef __attribute__((ext_vector_type(8)))  float    v8f;

#define NEDGE   512
#define NPTS    128
#define NNODES  8192

// ---------------------------------------------------------------------------
// WMMA helpers (CDNA5: v_wmma_f32_16x16x32_f16, f32 accumulate)
// ---------------------------------------------------------------------------
__device__ __forceinline__ v8f wmma_f16(v16h a, v16h b, v8f c) {
    return __builtin_amdgcn_wmma_f32_16x16x32_f16(false, a, false, b,
                                                  (short)0, c, false, false);
}

// Load a 16x32 f16 fragment (A layout; also used for B from an [n][k] panel).
// lane L: row = L%16, k runs [8*(L/16), +8) and [16+8*(L/16), +8).
__device__ __forceinline__ v16h load_frag(const h16* base, int ld) {
    int lane = threadIdx.x & 31;
    const h16* p = base + (size_t)(lane & 15) * ld + ((lane >> 4) << 3);
    v8h lo = *(const v8h*)p;
    v8h hi = *(const v8h*)(p + 16);
    v16h r;
#pragma unroll
    for (int i = 0; i < 8; i++) { r[i] = lo[i]; r[i + 8] = hi[i]; }
    return r;
}

// ---------------------------------------------------------------------------
// One-time weight packing: W f32 (Wrows x Nout, row-major) -> Wp f16 [Npad][Kpad]
// (n-major, k contiguous), zero padded. Amortized across all consuming blocks.
// ---------------------------------------------------------------------------
__global__ void pack_w_kernel(const float* __restrict__ W, h16* __restrict__ Wp,
                              int Wrows, int Nout, int Npad, int Kpad)
{
    long long i = (long long)blockIdx.x * blockDim.x + threadIdx.x;
    long long total = (long long)Npad * Kpad;
    if (i >= total) return;
    int n = (int)(i / Kpad), k = (int)(i % Kpad);
    Wp[i] = (h16)((n < Nout && k < Wrows) ? W[(size_t)k * Nout + n] : 0.f);
}

// ---------------------------------------------------------------------------
// Block-level GEMM: A (MT*16 x K) f16 in LDS, Wp f16 packed [N][K] in global
// (L2-resident). No weight staging, no per-K barriers. 256 threads (8 waves);
// wave w owns column tile w of each 128-wide chunk.
// MODE 0: out16 LDS f16 [M][ldo]; MODE 1: out32 LDS f32; MODE 2: column max
// over all MT*16 rows -> global f32/f16 (g-MLP max-pool).
// ---------------------------------------------------------------------------
template<int MT, int MODE>
__device__ void block_gemm(const h16* Alds, int lda, int K,
                           const h16* __restrict__ Wp, int N,
                           const float* __restrict__ bias, bool do_relu,
                           h16* out16, float* out32, int ldo,
                           float* gmax32, h16* gmax16)
{
    const int tid  = threadIdx.x;
    const int w    = tid >> 5;
    const int lane = tid & 31;
    for (int nc = 0; nc < N; nc += 128) {
        const int chunkN = (N - nc < 128) ? (N - nc) : 128;
        const int ctn    = chunkN >> 4;
        if (w < ctn) {
            v8f acc[MT];
            v8f zz = {};
#pragma unroll
            for (int i = 0; i < MT; i++) acc[i] = zz;
            const h16* wbase = Wp + (size_t)(nc + w * 16) * K;
            for (int k0 = 0; k0 < K; k0 += 32) {
                if (lane == 0 && k0 + 32 < K)
                    __builtin_prefetch(wbase + (size_t)k0 + 32, 0, 1);
                v16h bf = load_frag(wbase + k0, K);
#pragma unroll
                for (int rt = 0; rt < MT; rt++) {
                    v16h af = load_frag(Alds + (size_t)(rt * 16) * lda + k0, lda);
                    acc[rt] = wmma_f16(af, bf, acc[rt]);
                }
            }
            int nl = w * 16 + (lane & 15);
            int gn = nc + nl;
            float bv = bias ? bias[gn] : 0.f;
            if (MODE == 2) {
                float mx = -3.0e38f;
#pragma unroll
                for (int rt = 0; rt < MT; rt++) {
#pragma unroll
                    for (int r = 0; r < 8; r++) {
                        float v = acc[rt][r] + bv;
                        if (do_relu) v = fmaxf(v, 0.f);
                        mx = fmaxf(mx, v);
                    }
                }
                mx = fmaxf(mx, __shfl_xor(mx, 16, 32));
                if (lane < 16) {
                    if (gmax32) gmax32[gn] = mx;
                    if (gmax16) gmax16[gn] = (h16)mx;
                }
            } else {
#pragma unroll
                for (int rt = 0; rt < MT; rt++) {
#pragma unroll
                    for (int r = 0; r < 8; r++) {
                        int m = rt * 16 + r + ((lane >> 4) << 3);
                        float v = acc[rt][r] + bv;
                        if (do_relu) v = fmaxf(v, 0.f);
                        if (MODE == 0) out16[(size_t)m * ldo + gn] = (h16)v;
                        else           out32[(size_t)m * ldo + gn] = v;
                    }
                }
            }
        }
    }
    __syncthreads();   // make LDS outputs visible to the next stage
}

// ---------------------------------------------------------------------------
// Generic global GEMM: C = act(A(f16,MxK) @ Wp(f16 packed [Npad][K]) + b).
// Pure-register kernel: direct global fragment loads, no LDS, no barriers.
// 64x64 tile, 128 threads (4 waves). K mult of 32, M mult of 64, Nout guarded.
// ---------------------------------------------------------------------------
__global__ __launch_bounds__(128)
void gemm_f16_kernel(const h16* __restrict__ A, const h16* __restrict__ Wp,
                     const float* __restrict__ bias,
                     float* __restrict__ C32, h16* __restrict__ C16,
                     int M, int Nout, int K, int do_relu)
{
    int tid = threadIdx.x, w = tid >> 5, lane = tid & 31;
    int m0 = blockIdx.x * 64, n0 = blockIdx.y * 64;
    const h16* abase = A + (size_t)(m0 + w * 16) * K;
    v8f acc[4];
    v8f zz = {};
#pragma unroll
    for (int i = 0; i < 4; i++) acc[i] = zz;

    for (int k0 = 0; k0 < K; k0 += 32) {
        if (lane == 0 && k0 + 32 < K) {
            __builtin_prefetch(abase + (size_t)k0 + 32, 0, 1);
            __builtin_prefetch(Wp + (size_t)(n0 + w * 16) * K + k0 + 32, 0, 1);
        }
        v16h af = load_frag(abase + k0, K);
#pragma unroll
        for (int nt = 0; nt < 4; nt++) {
            v16h bf = load_frag(Wp + (size_t)(n0 + nt * 16) * K + k0, K);
            acc[nt] = wmma_f16(af, bf, acc[nt]);
        }
    }
#pragma unroll
    for (int nt = 0; nt < 4; nt++) {
        int gn = n0 + nt * 16 + (lane & 15);
        if (gn < Nout) {
            float bv = bias ? bias[gn] : 0.f;
#pragma unroll
            for (int r = 0; r < 8; r++) {
                int gm = m0 + w * 16 + r + ((lane >> 4) << 3);
                float v = acc[nt][r] + bv;
                if (do_relu) v = fmaxf(v, 0.f);
                if (C32) C32[(size_t)gm * Nout + gn] = v;
                if (C16) C16[(size_t)gm * Nout + gn] = (h16)v;
            }
        }
    }
}

// ---------------------------------------------------------------------------
// Farthest point sampling: one block per edge, np threads (np power of 2).
// ---------------------------------------------------------------------------
__global__ void fps_kernel(const float* __restrict__ Pg, float* __restrict__ outPos,
                           int np, int m)
{
    __shared__ float rv[128];
    __shared__ int   ri[128];
    __shared__ float cur[6];
    int e = blockIdx.x, t = threadIdx.x;
    const float* P = Pg + (size_t)e * np * 6;
    float p[6];
#pragma unroll
    for (int d = 0; d < 6; d++) p[d] = P[t * 6 + d];
    if (t == 0) {
#pragma unroll
        for (int d = 0; d < 6; d++) { cur[d] = p[d]; outPos[(size_t)e * m * 6 + d] = p[d]; }
    }
    __syncthreads();
    float mind = 0.f;
#pragma unroll
    for (int d = 0; d < 6; d++) { float df = p[d] - cur[d]; mind += df * df; }
    for (int i = 1; i < m; i++) {
        rv[t] = mind; ri[t] = t;
        __syncthreads();
        for (int s = np >> 1; s > 0; s >>= 1) {
            if (t < s) {
                float ov = rv[t + s]; int oi = ri[t + s];
                if (ov > rv[t] || (ov == rv[t] && oi < ri[t])) { rv[t] = ov; ri[t] = oi; }
            }
            __syncthreads();
        }
        int nxt = ri[0];
        if (t == nxt) {
#pragma unroll
            for (int d = 0; d < 6; d++) { cur[d] = p[d]; outPos[(size_t)e * m * 6 + i * 6 + d] = p[d]; }
        }
        __syncthreads();
        float dd = 0.f;
#pragma unroll
        for (int d = 0; d < 6; d++) { float df = p[d] - cur[d]; dd += df * df; }
        mind = fminf(mind, dd);
    }
}

// ---------------------------------------------------------------------------
// Radius k-NN (k=16): lexicographic selection of the 16 smallest (d2, idx)
// pairs with d2 <= r2. One block per edge, one thread per query.
// ---------------------------------------------------------------------------
__global__ void radius_kernel(const float* __restrict__ Qg, const float* __restrict__ Pg,
                              int nq, int np, float r2,
                              int* __restrict__ nbr, int* __restrict__ val)
{
    __shared__ float sp[128 * 6];
    int e = blockIdx.x, t = threadIdx.x;
    const float* P = Pg + (size_t)e * np * 6;
    for (int i = t; i < np * 6; i += nq) sp[i] = P[i];
    __syncthreads();
    float q[6];
    const float* Q = Qg + ((size_t)e * nq + t) * 6;
#pragma unroll
    for (int d = 0; d < 6; d++) q[d] = Q[d];
    float lastD = -1.f; int lastI = -1;
    for (int slot = 0; slot < 16; slot++) {
        float best = 3e38f; int bi = -1;
        for (int pi = 0; pi < np; pi++) {
            float d2 = 0.f;
#pragma unroll
            for (int d = 0; d < 6; d++) { float df = q[d] - sp[pi * 6 + d]; d2 += df * df; }
            if (d2 > r2) continue;
            if (d2 < lastD || (d2 == lastD && pi <= lastI)) continue;
            if (d2 < best || (d2 == best && pi < bi)) { best = d2; bi = pi; }
        }
        size_t o = ((size_t)e * nq + t) * 16 + slot;
        if (bi >= 0) { nbr[o] = bi; val[o] = 1; lastD = best; lastI = bi; }
        else         { nbr[o] = 0;  val[o] = 0; }
    }
}

// ---------------------------------------------------------------------------
// sa1 fused: per (edge, m1) block. rel(16x6) -> 64 -> 64 -> 128, masked max
// over the 16 neighbors -> x1 row (f16). Weights pre-packed f16.
// ---------------------------------------------------------------------------
__global__ __launch_bounds__(256)
void sa1_kernel(const float* __restrict__ synapse, const float* __restrict__ pos1,
                const int* __restrict__ nbr1, const int* __restrict__ val1,
                const float* W1, const float* b1,
                const h16* __restrict__ pW2, const float* b2,
                const h16* __restrict__ pW3, const float* b3,
                h16* __restrict__ x1f)
{
    __shared__ int   nb[16], vf[16];
    __shared__ float center[6];
    __shared__ float rel[16 * 6];
    __shared__ h16   A1[16 * 64];
    __shared__ h16   A2[16 * 64];
    __shared__ float H3[16 * 128];
    int e = blockIdx.x >> 6, m = blockIdx.x & 63;
    int t = threadIdx.x;
    size_t row = (size_t)e * 64 + m;
    if (t < 6)  center[t] = pos1[row * 6 + t];
    if (t < 16) { nb[t] = nbr1[row * 16 + t]; vf[t] = val1[row * 16 + t]; }
    __syncthreads();
    if (t < 96) {
        int j = t / 6, d = t % 6;
        rel[t] = synapse[((size_t)e * NPTS + nb[j]) * 6 + d] - center[d];
    }
    __syncthreads();
    for (int i = t; i < 16 * 64; i += 256) {
        int j = i >> 6, n = i & 63;
        float s = b1[n];
#pragma unroll
        for (int d = 0; d < 6; d++) s += rel[j * 6 + d] * W1[d * 64 + n];
        A1[i] = (h16)fmaxf(s, 0.f);
    }
    __syncthreads();
    block_gemm<1, 0>(A1, 64, 64, pW2, 64,  b2, true,  A2, nullptr, 64,  nullptr, nullptr);
    block_gemm<1, 1>(A2, 64, 64, pW3, 128, b3, false, nullptr, H3, 128, nullptr, nullptr);
    for (int c = t; c < 128; c += 256) {
        float mx = -1e30f;
#pragma unroll
        for (int j = 0; j < 16; j++) if (vf[j]) mx = fmaxf(mx, H3[j * 128 + c]);
        x1f[row * 128 + c] = (h16)mx;
    }
}

// ---------------------------------------------------------------------------
// Point-transformer attention, fused per (edge, m2) block.
// ---------------------------------------------------------------------------
__global__ __launch_bounds__(256)
void pt_attn_kernel(const float* __restrict__ pos1,
                    const int* __restrict__ nbr2, const int* __restrict__ val2,
                    const float* pn_W1, const float* pn_b1,
                    const h16* __restrict__ pPn2, const float* pn_b2,
                    const h16* __restrict__ pAttn, const float* attn_b,
                    const h16* __restrict__ asrc, const h16* __restrict__ adst,
                    const h16* __restrict__ vv, const float* __restrict__ logit_s,
                    const float* __restrict__ delta0, h16* __restrict__ x2f)
{
    __shared__ int   nb[16], vf[16];
    __shared__ float center[6];
    __shared__ float rel[16 * 6];
    __shared__ h16   O1[16 * 64];
    __shared__ float DLT[16 * 256];
    __shared__ h16   LA[16 * 256];
    __shared__ float LG[16 * 256];
    int e = blockIdx.x >> 4, m = blockIdx.x & 15;
    int t = threadIdx.x;
    size_t row = (size_t)e * 64 + m;
    if (t < 6)  center[t] = pos1[row * 6 + t];
    if (t < 16) { nb[t] = nbr2[((size_t)e * 16 + m) * 16 + t]; vf[t] = val2[((size_t)e * 16 + m) * 16 + t]; }
    __syncthreads();
    if (t < 96) {
        int j = t / 6, d = t % 6;
        rel[t] = center[d] - pos1[((size_t)e * 64 + nb[j]) * 6 + d];
    }
    __syncthreads();
    for (int i = t; i < 16 * 64; i += 256) {
        int j = i >> 6, n = i & 63;
        float s = pn_b1[n];
#pragma unroll
        for (int d = 0; d < 6; d++) s += rel[j * 6 + d] * pn_W1[d * 64 + n];
        O1[i] = (h16)fmaxf(s, 0.f);
    }
    __syncthreads();
    block_gemm<1, 1>(O1, 64, 64, pPn2, 256, pn_b2, true, nullptr, DLT, 256, nullptr, nullptr);
    for (int i = t; i < 16 * 256; i += 256) {
        int j = i >> 8, c = i & 255;
        float v = (float)adst[row * 256 + c]
                - (float)asrc[((size_t)e * 64 + nb[j]) * 256 + c] + DLT[i];
        LA[i] = (h16)v;
    }
    __syncthreads();
    block_gemm<1, 1>(LA, 256, 256, pAttn, 256, attn_b, false, nullptr, LG, 256, nullptr, nullptr);
    {
        int c = t;
        float ls = logit_s[row * 256 + c];
        float mx = ls;
#pragma unroll
        for (int j = 0; j < 16; j++) if (vf[j]) mx = fmaxf(mx, LG[j * 256 + c]);
        float es = __expf(ls - mx);
        float dsum = es;
        float msum = es * ((float)vv[row * 256 + c] + delta0[c]);
        for (int j = 0; j < 16; j++) {
            if (!vf[j]) continue;
            float ej = __expf(LG[j * 256 + c] - mx);
            dsum += ej;
            msum += ej * ((float)vv[((size_t)e * 64 + nb[j]) * 256 + c] + DLT[j * 256 + c]);
        }
        x2f[row * 256 + c] = (h16)(msum / dsum);
    }
}

// ---------------------------------------------------------------------------
// g-MLP fused per edge: concat(x2, pos1) 262(->288 pad) -> 256 -> 512 -> 1024,
// column max over the 64 points -> g (f16). Dynamic LDS (~132 KB of 320 KB).
// ---------------------------------------------------------------------------
__global__ __launch_bounds__(256)
void g_fused_kernel(const h16* __restrict__ x2f, const float* __restrict__ pos1,
                    const h16* __restrict__ pW1, const float* b1,
                    const h16* __restrict__ pW2, const float* b2,
                    const h16* __restrict__ pW3, const float* b3,
                    h16* __restrict__ gf16)
{
    extern __shared__ __align__(16) char smem[];
    h16* A0 = (h16*)smem;              // 64 x 288
    h16* H1 = A0 + 64 * 288;           // 64 x 256
    h16* H2 = H1 + 64 * 256;           // 64 x 512
    int e = blockIdx.x, t = threadIdx.x;
    for (int i = t; i < 64 * 288; i += 256) {
        int mr = i / 288, c = i - mr * 288;
        float v;
        if (c < 256)      v = (float)x2f[((size_t)e * 64 + mr) * 256 + c];
        else if (c < 262) v = pos1[((size_t)e * 64 + mr) * 6 + (c - 256)];
        else              v = 0.f;
        A0[i] = (h16)v;
    }
    __syncthreads();
    block_gemm<4, 0>(A0, 288, 288, pW1, 256,  b1, true,  H1, nullptr, 256, nullptr, nullptr);
    block_gemm<4, 0>(H1, 256, 256, pW2, 512,  b2, true,  H2, nullptr, 512, nullptr, nullptr);
    block_gemm<4, 2>(H2, 512, 512, pW3, 1024, b3, false, nullptr, nullptr, 0,
                     nullptr, gf16 + (size_t)e * 1024);
}

// ---------------------------------------------------------------------------
// Small elementwise / scatter / GAT kernels
// ---------------------------------------------------------------------------
__device__ __forceinline__ float leakyf(float x) { return x > 0.f ? x : 0.2f * x; }

__device__ __forceinline__ void atomicMaxF(float* addr, float v) {
    unsigned int* ua = (unsigned int*)addr;
    unsigned int old = *ua;
    while (__uint_as_float(old) < v) {
        unsigned int assumed = old;
        old = atomicCAS(ua, assumed, __float_as_uint(v));
        if (old == assumed) break;
    }
}

__global__ void fill_kernel(float* p, long long n, float v) {
    long long i = (long long)blockIdx.x * blockDim.x + threadIdx.x;
    if (i < n) p[i] = v;
}

__global__ void delta0_kernel(const float* b1, const float* W2, const float* b2, float* d0) {
    __shared__ float h1[64];
    int t = threadIdx.x;
    if (t < 64) h1[t] = fmaxf(b1[t], 0.f);
    __syncthreads();
    if (t < 256) {
        float s = b2[t];
        for (int j = 0; j < 64; j++) s += h1[j] * W2[j * 256 + t];
        d0[t] = fmaxf(s, 0.f);
    }
}

__global__ void tmpa_kernel(const h16* adst, const h16* asrc, const float* d0,
                            h16* out, long long n) {
    long long i = (long long)blockIdx.x * blockDim.x + threadIdx.x;
    if (i < n) out[i] = (h16)((float)adst[i] - (float)asrc[i] + d0[i & 255]);
}

__global__ void msgs_kernel(const h16* vv, const float* d0, h16* x2f, long long n) {
    long long i = (long long)blockIdx.x * blockDim.x + threadIdx.x;
    if (i < n) x2f[i] = (h16)((float)vv[i] + d0[i & 255]);
}

__global__ void conv16_kernel(const float* in, h16* out, long long n) {
    long long i = (long long)blockIdx.x * blockDim.x + threadIdx.x;
    if (i < n) out[i] = (h16)in[i];
}

__global__ void scatter_add_kernel(const int* ei, const float* xp,
                                   float* left, float* right, float* cl, float* cr) {
    int e = blockIdx.x, t = threadIdx.x;
    int s = ei[e], d = ei[NEDGE + e];
    float v = xp[(size_t)e * 512 + t];
    atomicAdd(&left[(size_t)s * 512 + t], v);
    atomicAdd(&right[(size_t)d * 512 + t], v);
    if (t == 0) { atomicAdd(&cl[s], 1.f); atomicAdd(&cr[d], 1.f); }
}

__global__ void scatter_div_kernel(float* buf, const float* cnt, long long n) {
    long long i = (long long)blockIdx.x * blockDim.x + threadIdx.x;
    if (i < n) buf[i] /= fmaxf(cnt[i / 512], 1.f);
}

__global__ void gat_scores_kernel(const float* xp, const float* a, float* s,
                                  int n, int H, int C) {
    int i = blockIdx.x * blockDim.x + threadIdx.x;
    if (i >= n * H) return;
    int node = i / H, h = i % H;
    float acc = 0.f;
    for (int c = 0; c < C; c++) acc += xp[(size_t)node * H * C + h * C + c] * a[h * C + c];
    s[i] = acc;
}

__global__ void gat_self_init_kernel(const float* ssc, const float* dsc, float* mx, int n) {
    int i = blockIdx.x * blockDim.x + threadIdx.x;
    if (i < n) mx[i] = leakyf(ssc[i] + dsc[i]);
}

__global__ void gat_edge_max_kernel(const int* ei, const float* ssc, const float* dsc,
                                    float* mx, int E, int H) {
    int i = blockIdx.x * blockDim.x + threadIdx.x;
    if (i >= E * H) return;
    int e = i / H, h = i % H;
    int s = ei[e], d = ei[E + e];
    atomicMaxF(&mx[d * H + h], leakyf(ssc[s * H + h] + dsc[d * H + h]));
}

__global__ void gat_self_z_kernel(const float* ssc, const float* dsc, const float* mx,
                                  float* z, int n) {
    int i = blockIdx.x * blockDim.x + threadIdx.x;
    if (i < n) z[i] = __expf(leakyf(ssc[i] + dsc[i]) - mx[i]);
}

__global__ void gat_edge_z_kernel(const int* ei, const float* ssc, const float* dsc,
                                  const float* mx, float* z, int E, int H) {
    int i = blockIdx.x * blockDim.x + threadIdx.x;
    if (i >= E * H) return;
    int e = i / H, h = i % H;
    int s = ei[e], d = ei[E + e];
    atomicAdd(&z[d * H + h], __expf(leakyf(ssc[s * H + h] + dsc[d * H + h]) - mx[d * H + h]));
}

__global__ void gat_edge_acc_kernel(const int* ei, const float* ssc, const float* dsc,
                                    const float* mx, const float* z, const float* xp,
                                    float* out, int E, int H, int C) {
    int i = blockIdx.x * blockDim.x + threadIdx.x;
    if (i >= E * H) return;
    int e = i / H, h = i % H;
    int s = ei[e], d = ei[E + e];
    float alpha = __expf(leakyf(ssc[s * H + h] + dsc[d * H + h]) - mx[d * H + h]) / z[d * H + h];
    size_t ob = (size_t)d * H * C + h * C, sb = (size_t)s * H * C + h * C;
    for (int c = 0; c < C; c++) atomicAdd(&out[ob + c], alpha * xp[sb + c]);
}

__global__ void gat_self_acc_kernel(const float* ssc, const float* dsc,
                                    const float* mx, const float* z, const float* xp,
                                    float* out, int n, int H, int C) {
    int i = blockIdx.x * blockDim.x + threadIdx.x;
    if (i >= n * H) return;
    float alpha = __expf(leakyf(ssc[i] + dsc[i]) - mx[i]) / z[i];
    int node = i / H, h = i % H;
    size_t b = (size_t)node * H * C + h * C;
    for (int c = 0; c < C; c++) out[b + c] += alpha * xp[b + c];
}

__global__ void gat1_fin_kernel(const float* out1, const float* b, h16* hg, long long n) {
    long long i = (long long)blockIdx.x * blockDim.x + threadIdx.x;
    if (i >= n) return;
    float v = out1[i] + b[i % 512];
    v = v > 0.f ? v : (__expf(v) - 1.f);   // ELU
    hg[i] = (h16)v;
}

__global__ void gat2_fin_kernel(const float* out2, const float* b, float* cache, long long n) {
    long long i = (long long)blockIdx.x * blockDim.x + threadIdx.x;
    if (i < n) cache[i] = out2[i] + b[i % 512];
}

__global__ void concat_kernel(const float* cache, const float* left, const float* right,
                              h16* f, long long n) {
    long long i = (long long)blockIdx.x * blockDim.x + threadIdx.x;
    if (i >= n) return;
    long long node = i / 1536; int j = (int)(i % 1536);
    float v = (j < 512) ? cache[node * 512 + j]
            : (j < 1024) ? left[node * 512 + (j - 512)]
                         : right[node * 512 + (j - 1024)];
    f[i] = (h16)v;
}

// ---------------------------------------------------------------------------
// Host orchestration
// ---------------------------------------------------------------------------
static inline void pack_w(hipStream_t stream, const float* W, h16* Wp,
                          int Wrows, int Nout, int Npad, int Kpad)
{
    long long total = (long long)Npad * Kpad;
    pack_w_kernel<<<(unsigned)((total + 255) / 256), 256, 0, stream>>>(W, Wp, Wrows, Nout, Npad, Kpad);
}

extern "C" void kernel_launch(void* const* d_in, const int* in_sizes, int n_in,
                              void* d_out, int out_size, void* d_ws, size_t ws_size,
                              hipStream_t stream)
{
    (void)in_sizes; (void)n_in; (void)out_size; (void)ws_size;
    const int*   edge_index = (const int*)  d_in[0];
    const float* synapse    = (const float*)d_in[2];
    const float* x_param    = (const float*)d_in[5];
    const float* gat1_W  = (const float*)d_in[6];
    const float* gat1_as = (const float*)d_in[7];
    const float* gat1_ad = (const float*)d_in[8];
    const float* gat1_b  = (const float*)d_in[9];
    const float* gat2_W  = (const float*)d_in[10];
    const float* gat2_as = (const float*)d_in[11];
    const float* gat2_ad = (const float*)d_in[12];
    const float* gat2_b  = (const float*)d_in[13];
    const float* sa1_W1 = (const float*)d_in[14];
    const float* sa1_b1 = (const float*)d_in[15];
    const float* sa1_W2 = (const float*)d_in[16];
    const float* sa1_b2 = (const float*)d_in[17];
    const float* sa1_W3 = (const float*)d_in[18];
    const float* sa1_b3 = (const float*)d_in[19];
    const float* pt_Wv   = (const float*)d_in[20];
    const float* pt_Wsrc = (const float*)d_in[21];
    const float* pt_Wdst = (const float*)d_in[22];
    const float* pn_W1 = (const float*)d_in[23];
    const float* pn_b1 = (const float*)d_in[24];
    const float* pn_W2 = (const float*)d_in[25];
    const float* pn_b2 = (const float*)d_in[26];
    const float* attn_W = (const float*)d_in[27];
    const float* attn_b = (const float*)d_in[28];
    const float* g_W1 = (const float*)d_in[29];
    const float* g_b1 = (const float*)d_in[30];
    const float* g_W2 = (const float*)d_in[31];
    const float* g_b2 = (const float*)d_in[32];
    const float* g_W3 = (const float*)d_in[33];
    const float* g_b3 = (const float*)d_in[34];
    const float* enc_W1 = (const float*)d_in[35];
    const float* enc_b1 = (const float*)d_in[36];
    const float* enc_W2 = (const float*)d_in[37];
    const float* enc_b2 = (const float*)d_in[38];
    const float* cls_W1 = (const float*)d_in[39];
    const float* cls_b1 = (const float*)d_in[40];
    const float* cls_W2 = (const float*)d_in[41];
    const float* cls_b2 = (const float*)d_in[42];
    float* out = (float*)d_out;

    char* base = (char*)d_ws; size_t off = 0;
    auto alloc = [&](size_t bytes) -> void* {
        void* p = base + off; off = (off + bytes + 255) & ~(size_t)255; return p;
    };
    // activations / intermediates
    float* pos1 = (float*)alloc((size_t)NEDGE * 64 * 6 * 4);
    float* q2   = (float*)alloc((size_t)NEDGE * 16 * 6 * 4);
    int* nbr1 = (int*)alloc((size_t)NEDGE * 64 * 16 * 4);
    int* val1 = (int*)alloc((size_t)NEDGE * 64 * 16 * 4);
    int* nbr2 = (int*)alloc((size_t)NEDGE * 16 * 16 * 4);
    int* val2 = (int*)alloc((size_t)NEDGE * 16 * 16 * 4);
    h16* x1f  = (h16*)alloc((size_t)32768 * 128 * 2);
    h16* asrc = (h16*)alloc((size_t)32768 * 256 * 2);
    h16* adst = (h16*)alloc((size_t)32768 * 256 * 2);
    h16* vv   = (h16*)alloc((size_t)32768 * 256 * 2);
    float* d0 = (float*)alloc(256 * 4);
    h16* tmpA = (h16*)alloc((size_t)32768 * 256 * 2);
    float* lgs = (float*)alloc((size_t)32768 * 256 * 4);
    h16* x2f  = (h16*)alloc((size_t)32768 * 256 * 2);
    h16* gf   = (h16*)alloc((size_t)NEDGE * 1024 * 2);
    h16* ench = (h16*)alloc((size_t)NEDGE * 512 * 2);
    float* xpt = (float*)alloc((size_t)NEDGE * 512 * 4);
    float* left  = (float*)alloc((size_t)NNODES * 512 * 4);
    float* right = (float*)alloc((size_t)NNODES * 512 * 4);
    float* cl = (float*)alloc((size_t)NNODES * 4);
    float* cr = (float*)alloc((size_t)NNODES * 4);
    h16* xp16 = (h16*)alloc((size_t)NNODES * 128 * 2);
    float* xp1 = (float*)alloc((size_t)NNODES * 512 * 4);
    float* ssc1 = (float*)alloc((size_t)NNODES * 8 * 4);
    float* dsc1 = (float*)alloc((size_t)NNODES * 8 * 4);
    float* mx1  = (float*)alloc((size_t)NNODES * 8 * 4);
    float* z1   = (float*)alloc((size_t)NNODES * 8 * 4);
    float* out1 = (float*)alloc((size_t)NNODES * 512 * 4);
    h16* hg   = (h16*)alloc((size_t)NNODES * 512 * 2);
    float* xp2  = (float*)alloc((size_t)NNODES * 512 * 4);
    float* ssc2 = (float*)alloc((size_t)NNODES * 4);
    float* dsc2 = (float*)alloc((size_t)NNODES * 4);
    float* mx2  = (float*)alloc((size_t)NNODES * 4);
    float* z2   = (float*)alloc((size_t)NNODES * 4);
    float* out2 = (float*)alloc((size_t)NNODES * 512 * 4);
    float* cache = (float*)alloc((size_t)NNODES * 512 * 4);
    h16* fcat = (h16*)alloc((size_t)NNODES * 1536 * 2);
    h16* clsh = (h16*)alloc((size_t)NNODES * 512 * 2);
    // packed weights (f16, [Npad][Kpad])
    h16* pWsrc = (h16*)alloc((size_t)256 * 128 * 2);
    h16* pWdst = (h16*)alloc((size_t)256 * 128 * 2);
    h16* pWv   = (h16*)alloc((size_t)256 * 128 * 2);
    h16* pAttn = (h16*)alloc((size_t)256 * 256 * 2);
    h16* pSa2  = (h16*)alloc((size_t)64 * 64 * 2);
    h16* pSa3  = (h16*)alloc((size_t)128 * 64 * 2);
    h16* pPn2  = (h16*)alloc((size_t)256 * 64 * 2);
    h16* pG1   = (h16*)alloc((size_t)256 * 288 * 2);
    h16* pG2   = (h16*)alloc((size_t)512 * 256 * 2);
    h16* pG3   = (h16*)alloc((size_t)1024 * 512 * 2);
    h16* pE1   = (h16*)alloc((size_t)512 * 1024 * 2);
    h16* pE2   = (h16*)alloc((size_t)512 * 512 * 2);
    h16* pGat1 = (h16*)alloc((size_t)512 * 128 * 2);
    h16* pGat2 = (h16*)alloc((size_t)512 * 512 * 2);
    h16* pCls1 = (h16*)alloc((size_t)512 * 1536 * 2);
    h16* pCls2 = (h16*)alloc((size_t)192 * 512 * 2);

    // ---- pack all weights once (amortized over thousands of blocks) ----
    pack_w(stream, pt_Wsrc, pWsrc, 128, 256, 256, 128);
    pack_w(stream, pt_Wdst, pWdst, 128, 256, 256, 128);
    pack_w(stream, pt_Wv,   pWv,   128, 256, 256, 128);
    pack_w(stream, attn_W,  pAttn, 256, 256, 256, 256);
    pack_w(stream, sa1_W2,  pSa2,  64,  64,  64,  64);
    pack_w(stream, sa1_W3,  pSa3,  64,  128, 128, 64);
    pack_w(stream, pn_W2,   pPn2,  64,  256, 256, 64);
    pack_w(stream, g_W1,    pG1,   262, 256, 256, 288);
    pack_w(stream, g_W2,    pG2,   256, 512, 512, 256);
    pack_w(stream, g_W3,    pG3,   512, 1024, 1024, 512);
    pack_w(stream, enc_W1,  pE1,   1024, 512, 512, 1024);
    pack_w(stream, enc_W2,  pE2,   512, 512, 512, 512);
    pack_w(stream, gat1_W,  pGat1, 128, 512, 512, 128);
    pack_w(stream, gat2_W,  pGat2, 512, 512, 512, 512);
    pack_w(stream, cls_W1,  pCls1, 1536, 512, 512, 1536);
    pack_w(stream, cls_W2,  pCls2, 512, 133, 192, 512);

    // ---- point branch ----
    fps_kernel<<<NEDGE, 128, 0, stream>>>(synapse, pos1, 128, 64);
    radius_kernel<<<NEDGE, 64, 0, stream>>>(pos1, synapse, 64, 128, 0.04f, nbr1, val1);
    fps_kernel<<<NEDGE, 64, 0, stream>>>(pos1, q2, 64, 16);
    radius_kernel<<<NEDGE, 16, 0, stream>>>(q2, pos1, 16, 64, 0.16f, nbr2, val2);
    sa1_kernel<<<32768, 256, 0, stream>>>(synapse, pos1, nbr1, val1,
                                          sa1_W1, sa1_b1, pSa2, sa1_b2, pSa3, sa1_b3, x1f);
    dim3 gproj(32768 / 64, 256 / 64);
    gemm_f16_kernel<<<gproj, 128, 0, stream>>>(x1f, pWsrc, nullptr, nullptr, asrc, 32768, 256, 128, 0);
    gemm_f16_kernel<<<gproj, 128, 0, stream>>>(x1f, pWdst, nullptr, nullptr, adst, 32768, 256, 128, 0);
    gemm_f16_kernel<<<gproj, 128, 0, stream>>>(x1f, pWv,   nullptr, nullptr, vv,   32768, 256, 128, 0);
    delta0_kernel<<<1, 256, 0, stream>>>(pn_b1, pn_W2, pn_b2, d0);
    long long tot = (long long)32768 * 256;
    tmpa_kernel<<<(unsigned)(tot / 256), 256, 0, stream>>>(adst, asrc, d0, tmpA, tot);
    gemm_f16_kernel<<<dim3(512, 4), 128, 0, stream>>>(tmpA, pAttn, attn_b, lgs, nullptr, 32768, 256, 256, 0);
    msgs_kernel<<<(unsigned)(tot / 256), 256, 0, stream>>>(vv, d0, x2f, tot);
    pt_attn_kernel<<<NEDGE * 16, 256, 0, stream>>>(pos1, nbr2, val2, pn_W1, pn_b1, pPn2, pn_b2,
                                                   pAttn, attn_b, asrc, adst, vv, lgs, d0, x2f);
    size_t g_smem = (size_t)64 * 288 * 2 + 64 * 256 * 2 + 64 * 512 * 2;
    g_fused_kernel<<<NEDGE, 256, g_smem, stream>>>(x2f, pos1, pG1, g_b1, pG2, g_b2, pG3, g_b3, gf);
    gemm_f16_kernel<<<dim3(8, 8), 128, 0, stream>>>(gf, pE1, enc_b1, nullptr, ench, 512, 512, 1024, 1);
    gemm_f16_kernel<<<dim3(8, 8), 128, 0, stream>>>(ench, pE2, enc_b2, xpt, nullptr, 512, 512, 512, 0);

    // ---- scatter mean ----
    long long nnc = (long long)NNODES * 512;
    fill_kernel<<<(unsigned)((nnc + 255) / 256), 256, 0, stream>>>(left, nnc, 0.f);
    fill_kernel<<<(unsigned)((nnc + 255) / 256), 256, 0, stream>>>(right, nnc, 0.f);
    fill_kernel<<<(NNODES + 255) / 256, 256, 0, stream>>>(cl, NNODES, 0.f);
    fill_kernel<<<(NNODES + 255) / 256, 256, 0, stream>>>(cr, NNODES, 0.f);
    scatter_add_kernel<<<NEDGE, 512, 0, stream>>>(edge_index, xpt, left, right, cl, cr);
    scatter_div_kernel<<<(unsigned)((nnc + 255) / 256), 256, 0, stream>>>(left, cl, nnc);
    scatter_div_kernel<<<(unsigned)((nnc + 255) / 256), 256, 0, stream>>>(right, cr, nnc);

    // ---- GAT layer 1 (8 heads x 64) ----
    conv16_kernel<<<(NNODES * 128) / 256, 256, 0, stream>>>(x_param, xp16, (long long)NNODES * 128);
    gemm_f16_kernel<<<dim3(128, 8), 128, 0, stream>>>(xp16, pGat1, nullptr, xp1, nullptr, NNODES, 512, 128, 0);
    gat_scores_kernel<<<(NNODES * 8 + 255) / 256, 256, 0, stream>>>(xp1, gat1_as, ssc1, NNODES, 8, 64);
    gat_scores_kernel<<<(NNODES * 8 + 255) / 256, 256, 0, stream>>>(xp1, gat1_ad, dsc1, NNODES, 8, 64);
    gat_self_init_kernel<<<(NNODES * 8 + 255) / 256, 256, 0, stream>>>(ssc1, dsc1, mx1, NNODES * 8);
    gat_edge_max_kernel<<<(NEDGE * 8 + 255) / 256, 256, 0, stream>>>(edge_index, ssc1, dsc1, mx1, NEDGE, 8);
    gat_self_z_kernel<<<(NNODES * 8 + 255) / 256, 256, 0, stream>>>(ssc1, dsc1, mx1, z1, NNODES * 8);
    gat_edge_z_kernel<<<(NEDGE * 8 + 255) / 256, 256, 0, stream>>>(edge_index, ssc1, dsc1, mx1, z1, NEDGE, 8);
    fill_kernel<<<(unsigned)((nnc + 255) / 256), 256, 0, stream>>>(out1, nnc, 0.f);
    gat_edge_acc_kernel<<<(NEDGE * 8 + 255) / 256, 256, 0, stream>>>(edge_index, ssc1, dsc1, mx1, z1, xp1, out1, NEDGE, 8, 64);
    gat_self_acc_kernel<<<(NNODES * 8 + 255) / 256, 256, 0, stream>>>(ssc1, dsc1, mx1, z1, xp1, out1, NNODES, 8, 64);
    gat1_fin_kernel<<<(unsigned)((nnc + 255) / 256), 256, 0, stream>>>(out1, gat1_b, hg, nnc);

    // ---- GAT layer 2 (1 head x 512) ----
    gemm_f16_kernel<<<dim3(128, 8), 128, 0, stream>>>(hg, pGat2, nullptr, xp2, nullptr, NNODES, 512, 512, 0);
    gat_scores_kernel<<<(NNODES + 255) / 256, 256, 0, stream>>>(xp2, gat2_as, ssc2, NNODES, 1, 512);
    gat_scores_kernel<<<(NNODES + 255) / 256, 256, 0, stream>>>(xp2, gat2_ad, dsc2, NNODES, 1, 512);
    gat_self_init_kernel<<<(NNODES + 255) / 256, 256, 0, stream>>>(ssc2, dsc2, mx2, NNODES);
    gat_edge_max_kernel<<<(NEDGE + 255) / 256, 256, 0, stream>>>(edge_index, ssc2, dsc2, mx2, NEDGE, 1);
    gat_self_z_kernel<<<(NNODES + 255) / 256, 256, 0, stream>>>(ssc2, dsc2, mx2, z2, NNODES);
    gat_edge_z_kernel<<<(NEDGE + 255) / 256, 256, 0, stream>>>(edge_index, ssc2, dsc2, mx2, z2, NEDGE, 1);
    fill_kernel<<<(unsigned)((nnc + 255) / 256), 256, 0, stream>>>(out2, nnc, 0.f);
    gat_edge_acc_kernel<<<(NEDGE + 255) / 256, 256, 0, stream>>>(edge_index, ssc2, dsc2, mx2, z2, xp2, out2, NEDGE, 1, 512);
    gat_self_acc_kernel<<<(NNODES + 255) / 256, 256, 0, stream>>>(ssc2, dsc2, mx2, z2, xp2, out2, NNODES, 1, 512);
    gat2_fin_kernel<<<(unsigned)((nnc + 255) / 256), 256, 0, stream>>>(out2, gat2_b, cache, nnc);

    // ---- classifier ----
    long long nf = (long long)NNODES * 1536;
    concat_kernel<<<(unsigned)((nf + 255) / 256), 256, 0, stream>>>(cache, left, right, fcat, nf);
    gemm_f16_kernel<<<dim3(128, 8), 128, 0, stream>>>(fcat, pCls1, cls_b1, nullptr, clsh, NNODES, 512, 1536, 1);
    gemm_f16_kernel<<<dim3(128, 3), 128, 0, stream>>>(clsh, pCls2, cls_b2, out, nullptr, NNODES, 133, 512, 0);
}
